// Attention_32478542693201
// MI455X (gfx1250) — compile-verified, matrix operand order checked by reading, not run on hardware
//
#include <hip/hip_runtime.h>

// ---------------- problem constants ----------------
#define T_SEQ 2048
#define DIM   4096
#define NH    32
#define NKV   8
#define HD    128
#define QKV_N ((NH + 2 * NKV) * HD) // 6144
#define KT    64                    // attention k-tile
#define QT    128                   // attention q-tile

typedef _Float16 half8   __attribute__((ext_vector_type(8)));
typedef _Float16 half16  __attribute__((ext_vector_type(16)));
typedef float    floatx8 __attribute__((ext_vector_type(8)));
typedef int      intx4   __attribute__((vector_size(16))); // builtin's pointee

// ---------------- CDNA5 async global->LDS copy (guarded) -------------------
#if defined(__has_builtin)
#if __has_builtin(__builtin_amdgcn_global_load_async_to_lds_b128)
#define HAS_ASYNC_LDS 1
#endif
#endif

// copy 16 bytes from global to LDS; async path is one ASYNCcnt instruction
static __device__ __forceinline__ void cp16_async(_Float16* l,
                                                  const _Float16* g) {
#ifdef HAS_ASYNC_LDS
  __builtin_amdgcn_global_load_async_to_lds_b128(
      (__attribute__((address_space(1))) intx4*)g,
      (__attribute__((address_space(3))) intx4*)l, 0, 0);
#else
  *(half8*)l = *(const half8*)g;
#endif
}

template <int N>
static __device__ __forceinline__ void wait_asynccnt() {
#ifdef HAS_ASYNC_LDS
#if __has_builtin(__builtin_amdgcn_s_wait_asynccnt)
  __builtin_amdgcn_s_wait_asynccnt(N);
#else
  asm volatile("s_wait_asynccnt %0" ::"i"(N) : "memory");
#endif
#endif
}

// D = A(16x32 f16) x B(32x16 f16) + C(16x16 f32)   -- wave32 CDNA5 WMMA
static __device__ __forceinline__ floatx8 wmma_f16(half16 a, half16 b,
                                                   floatx8 c) {
  return __builtin_amdgcn_wmma_f32_16x16x32_f16(false, a, false, b, (short)0, c,
                                                false, false);
}

// Per-lane fragment gather for 16-bit A/B operands (row-major source):
// lane l holds row (l&15); K-halves [c..c+7] and [c+16..c+23], c=(l>>4)*8.
static __device__ __forceinline__ half16 ldfrag(const _Float16* p) {
  half8 lo = *(const half8*)p;
  half8 hi = *(const half8*)(p + 16);
  return __builtin_shufflevector(lo, hi, 0, 1, 2, 3, 4, 5, 6, 7, 8, 9, 10, 11,
                                 12, 13, 14, 15);
}

// ---------------- kernel 0: f32 -> f16 weight conversion (once/call) -------
__global__ __launch_bounds__(256) void cvt_f32_f16(const float* __restrict__ s,
                                                   _Float16* __restrict__ d) {
  const size_t i = ((size_t)blockIdx.x * 256 + threadIdx.x) * 8;
  float4 a = *(const float4*)(s + i);
  float4 b = *(const float4*)(s + i + 4);
  half8 h;
  h[0] = (_Float16)a.x; h[1] = (_Float16)a.y;
  h[2] = (_Float16)a.z; h[3] = (_Float16)a.w;
  h[4] = (_Float16)b.x; h[5] = (_Float16)b.y;
  h[6] = (_Float16)b.z; h[7] = (_Float16)b.w;
  *(half8*)(d + i) = h;
}

// ---------------- kernel 1: per-row fake-quant of x ----------------
__global__ __launch_bounds__(256) void quant_rows(const float* __restrict__ X,
                                                  _Float16* __restrict__ Q,
                                                  float* __restrict__ inv_s) {
  __shared__ float red[256];
  const int t = blockIdx.x, tid = threadIdx.x;
  const float* xp = X + (size_t)t * DIM;
  float v[16], am = 0.f;
#pragma unroll
  for (int i = 0; i < 16; ++i) {
    v[i] = xp[tid + i * 256];
    am = fmaxf(am, fabsf(v[i]));
  }
  red[tid] = am;
  __syncthreads();
  for (int s = 128; s > 0; s >>= 1) {
    if (tid < s) red[tid] = fmaxf(red[tid], red[tid + s]);
    __syncthreads();
  }
  const float amax = fmaxf(red[0], 1e-5f);
  const float sc = 127.f / amax;
#pragma unroll
  for (int i = 0; i < 16; ++i) {
    float q = fminf(fmaxf(rintf(v[i] * sc), -128.f), 127.f);
    Q[(size_t)t * DIM + tid + i * 256] = (_Float16)q;
  }
  if (tid == 0) inv_s[t] = amax * (1.f / 127.f);
}

// ---------------- kernel 2/6: C[M,N] = (A_f16 @ B_f16^T) * inv_s[row] ------
// Block tile 128x256, 8 waves (2x4), 64x64 per wave = 16 WMMAs per k-step.
// Async double-buffered LDS staging: 6 b128 copies / thread / tile.
__global__ __launch_bounds__(256) void gemm_rowscale(
    const _Float16* __restrict__ A, const _Float16* __restrict__ B,
    const float* __restrict__ inv_s, float* __restrict__ C, int N, int K) {
  __shared__ alignas(16) _Float16 lA[2][128 * 40]; // 128 x (32+8 pad)
  __shared__ alignas(16) _Float16 lB[2][256 * 40]; // 256 x (32+8 pad)
  const int tid = threadIdx.x, lane = tid & 31, wave = tid >> 5;
  const int wm = wave & 1, wn = wave >> 1; // 2 x 4 wave grid
  const int m0 = blockIdx.y * 128, n0 = blockIdx.x * 256;
  const int rsel = lane & 15, chalf = lane >> 4, csel = chalf * 8;
  const int arow = tid >> 2, ach = (tid & 3) * 8;

  floatx8 acc[4][4] = {};

  auto stage = [&](int k0, int b) {
#pragma unroll
    for (int p = 0; p < 2; ++p)
      cp16_async(&lA[b][(arow + p * 64) * 40 + ach],
                 &A[(size_t)(m0 + arow + p * 64) * K + k0 + ach]);
#pragma unroll
    for (int p = 0; p < 4; ++p) {
      const int cid = tid + p * 256, r = cid >> 2, ch = (cid & 3) * 8;
      cp16_async(&lB[b][r * 40 + ch], &B[(size_t)(n0 + r) * K + k0 + ch]);
    }
  };

  const int nk = K / 32;
  stage(0, 0);
  for (int kt = 0; kt < nk; ++kt) {
    const int cur = kt & 1;
    if (kt + 1 < nk) {
      stage((kt + 1) * 32, cur ^ 1);
      wait_asynccnt<6>(); // retire tile kt's copies (in-order)
    } else {
      wait_asynccnt<0>();
    }
    __syncthreads();

    half16 af[4], bf[4];
#pragma unroll
    for (int i = 0; i < 4; ++i)
      af[i] = ldfrag(&lA[cur][(wm * 64 + i * 16 + rsel) * 40 + csel]);
#pragma unroll
    for (int j = 0; j < 4; ++j)
      bf[j] = ldfrag(&lB[cur][(wn * 64 + j * 16 + rsel) * 40 + csel]);
#pragma unroll
    for (int i = 0; i < 4; ++i)
#pragma unroll
      for (int j = 0; j < 4; ++j) acc[i][j] = wmma_f16(af[i], bf[j], acc[i][j]);
    __syncthreads(); // reads done before this buffer is re-staged
  }

#pragma unroll
  for (int i = 0; i < 4; ++i) {
#pragma unroll
    for (int r = 0; r < 8; ++r) {
      const int m = m0 + wm * 64 + i * 16 + chalf * 8 + r;
      const float sc = inv_s[m];
#pragma unroll
      for (int j = 0; j < 4; ++j) {
        const int n = n0 + wn * 64 + j * 16 + rsel;
        C[(size_t)m * N + n] = acc[i][j][r] * sc;
      }
    }
  }
}

// ---------------- kernel 3: RoPE + pack Q/K/V ------------------------------
// Qh[h][t][d], Kh[g][t][d]; Vt[g][d][t] transposed so PV's B-operand is
// row-major over the contraction (no transposes in the attention hot loop).
__global__ void rope_pack(const float* __restrict__ xqkv,
                          _Float16* __restrict__ Qh, _Float16* __restrict__ Kh,
                          _Float16* __restrict__ Vt) {
  const int idx = blockIdx.x * 256 + threadIdx.x; // T*3072 pair slots
  const int t = idx / (QKV_N / 2);
  const int n = (idx % (QKV_N / 2)) * 2;
  const float a = xqkv[(size_t)t * QKV_N + n];
  const float b = xqkv[(size_t)t * QKV_N + n + 1];
  if (n < NH * HD) { // Q
    const int h = n >> 7, d = n & 127, i = d >> 1;
    const float invf = __expf((float)i * -0.20503692627f); // theta^(-i/64)
    float s, c;
    __sincosf((float)t * invf, &s, &c);
    const size_t base = ((size_t)h * T_SEQ + t) * HD + d;
    Qh[base] = (_Float16)(a * c - b * s);
    Qh[base + 1] = (_Float16)(a * s + b * c);
  } else if (n < (NH + NKV) * HD) { // K
    const int m = n - NH * HD, g = m >> 7, d = m & 127, i = d >> 1;
    const float invf = __expf((float)i * -0.20503692627f);
    float s, c;
    __sincosf((float)t * invf, &s, &c);
    const size_t base = ((size_t)g * T_SEQ + t) * HD + d;
    Kh[base] = (_Float16)(a * c - b * s);
    Kh[base + 1] = (_Float16)(a * s + b * c);
  } else { // V (transposed store)
    const int m = n - (NH + NKV) * HD, g = m >> 7, d = m & 127;
    Vt[((size_t)g * HD + d) * T_SEQ + t] = (_Float16)a;
    Vt[((size_t)g * HD + d + 1) * T_SEQ + t] = (_Float16)b;
  }
}

// ---------------- kernel 4: flash attention --------------------------------
__global__ __launch_bounds__(256) void attn_fa(const _Float16* __restrict__ Qh,
                                               const _Float16* __restrict__ Kh,
                                               const _Float16* __restrict__ Vt,
                                               float* __restrict__ O) {
  __shared__ alignas(16) _Float16 lK[2][KT * 136];    // KT keys x (HD+8)
  __shared__ alignas(16) _Float16 lV[2][HD * 72];     // HD d-rows x (KT+8)
  __shared__ alignas(16) _Float16 lP[8 * 16 * 72];    // per-wave P staging
  const int tid = threadIdx.x, lane = tid & 31, wave = tid >> 5;
  const int h = blockIdx.y, grp = h >> 2;
  const int q0 = blockIdx.x * QT;
  const int rsel = lane & 15, chalf = lane >> 4, csel = chalf * 8;
  const int qw = q0 + wave * 16;
  const float scale = 0.08838834764831845f; // 1/sqrt(128)

  half16 qf[4];
  {
    const _Float16* qb = Qh + ((size_t)h * T_SEQ + qw + rsel) * HD;
#pragma unroll
    for (int kk = 0; kk < 4; ++kk) qf[kk] = ldfrag(qb + kk * 32 + csel);
  }

  floatx8 o[8] = {};
  float mrow[8], lrow[8];
#pragma unroll
  for (int r = 0; r < 8; ++r) {
    mrow[r] = -1e30f;
    lrow[r] = 0.f;
  }

  auto stage = [&](int s0, int b) {
#pragma unroll
    for (int c = 0; c < 4; ++c) { // K tile: KT x 128 halfs
      const int cid = tid + c * 256, row = cid >> 4, c8 = cid & 15;
      cp16_async(&lK[b][row * 136 + c8 * 8],
                 &Kh[((size_t)grp * T_SEQ + s0 + row) * HD + c8 * 8]);
    }
#pragma unroll
    for (int c = 0; c < 4; ++c) { // V^T tile: 128 x KT halfs
      const int cid = tid + c * 256, d = cid >> 3, c8 = cid & 7;
      cp16_async(&lV[b][d * 72 + c8 * 8],
                 &Vt[((size_t)grp * HD + d) * T_SEQ + s0 + c8 * 8]);
    }
  };

  const int nkt = q0 / KT + 2;
  stage(0, 0);
  for (int kt = 0; kt < nkt; ++kt) {
    const int s0 = kt * KT, cur = kt & 1;
    if (kt + 1 < nkt) {
      stage(s0 + KT, cur ^ 1);
      wait_asynccnt<8>();
    } else {
      wait_asynccnt<0>();
    }
    __syncthreads();

    // S = Q K^T
    floatx8 sacc[4] = {};
#pragma unroll
    for (int j = 0; j < 4; ++j)
#pragma unroll
      for (int kk = 0; kk < 4; ++kk)
        sacc[j] = wmma_f16(
            qf[kk], ldfrag(&lK[cur][(j * 16 + rsel) * 136 + kk * 32 + csel]),
            sacc[j]);

    // scale + causal mask
#pragma unroll
    for (int j = 0; j < 4; ++j) {
      const int col = s0 + j * 16 + rsel;
#pragma unroll
      for (int r = 0; r < 8; ++r) {
        float v = sacc[j][r] * scale;
        if (col > qw + chalf * 8 + r) v = -1e30f;
        sacc[j][r] = v;
      }
    }
    // online softmax (row lives in VGPR index x lane-half; 16-lane reduce)
#pragma unroll
    for (int r = 0; r < 8; ++r) {
      float lm = fmaxf(fmaxf(sacc[0][r], sacc[1][r]),
                       fmaxf(sacc[2][r], sacc[3][r]));
      for (int off = 1; off < 16; off <<= 1)
        lm = fmaxf(lm, __shfl_xor(lm, off, 32));
      const float mn = fmaxf(mrow[r], lm);
      const float alpha = __expf(mrow[r] - mn);
      mrow[r] = mn;
      float rs = 0.f;
#pragma unroll
      for (int j = 0; j < 4; ++j) {
        const float p = __expf(sacc[j][r] - mn);
        sacc[j][r] = p;
        rs += p;
      }
      for (int off = 1; off < 16; off <<= 1) rs += __shfl_xor(rs, off, 32);
      lrow[r] = lrow[r] * alpha + rs;
#pragma unroll
      for (int jd = 0; jd < 8; ++jd) o[jd][r] *= alpha;
    }
    // C-layout -> A-layout via per-wave LDS
    _Float16* pp = lP + wave * 16 * 72;
#pragma unroll
    for (int j = 0; j < 4; ++j)
#pragma unroll
      for (int r = 0; r < 8; ++r)
        pp[(chalf * 8 + r) * 72 + j * 16 + rsel] = (_Float16)sacc[j][r];
    // O += P V
#pragma unroll
    for (int kk = 0; kk < 2; ++kk) {
      const half16 pa = ldfrag(pp + rsel * 72 + kk * 32 + csel);
#pragma unroll
      for (int jd = 0; jd < 8; ++jd)
        o[jd] = wmma_f16(
            pa, ldfrag(&lV[cur][(jd * 16 + rsel) * 72 + kk * 32 + csel]),
            o[jd]);
    }
    __syncthreads(); // reads done before this buffer is re-staged
  }

#pragma unroll
  for (int r = 0; r < 8; ++r) {
    const float rl = 1.f / lrow[r];
    const int trow = qw + chalf * 8 + r;
#pragma unroll
    for (int jd = 0; jd < 8; ++jd)
      O[(size_t)trow * DIM + h * HD + jd * 16 + rsel] = o[jd][r] * rl;
  }
}

// ---------------- kernel 5: RMS-norm + fake-quant --------------------------
__global__ __launch_bounds__(256) void rms_quant_rows(
    const float* __restrict__ X, const float* __restrict__ rms_w,
    _Float16* __restrict__ Q, float* __restrict__ inv_s) {
  __shared__ float red[256];
  const int t = blockIdx.x, tid = threadIdx.x;
  const float* xp = X + (size_t)t * DIM;
  float v[16], ss = 0.f;
#pragma unroll
  for (int i = 0; i < 16; ++i) {
    v[i] = xp[tid + i * 256];
    ss += v[i] * v[i];
  }
  red[tid] = ss;
  __syncthreads();
  for (int s = 128; s > 0; s >>= 1) {
    if (tid < s) red[tid] += red[tid + s];
    __syncthreads();
  }
  const float rr = rsqrtf(red[0] * (1.f / DIM) + 1e-5f);
  __syncthreads();
  float am = 0.f;
#pragma unroll
  for (int i = 0; i < 16; ++i) {
    v[i] = v[i] * rr * rms_w[tid + i * 256];
    am = fmaxf(am, fabsf(v[i]));
  }
  red[tid] = am;
  __syncthreads();
  for (int s = 128; s > 0; s >>= 1) {
    if (tid < s) red[tid] = fmaxf(red[tid], red[tid + s]);
    __syncthreads();
  }
  const float amax = fmaxf(red[0], 1e-5f);
  const float sc = 127.f / amax;
#pragma unroll
  for (int i = 0; i < 16; ++i) {
    float q = fminf(fmaxf(rintf(v[i] * sc), -128.f), 127.f);
    Q[(size_t)t * DIM + tid + i * 256] = (_Float16)q;
  }
  if (tid == 0) inv_s[t] = amax * (1.f / 127.f);
}

// ---------------- host side ------------------------------------------------
extern "C" void kernel_launch(void* const* d_in, const int* in_sizes, int n_in,
                              void* d_out, int out_size, void* d_ws,
                              size_t ws_size, hipStream_t stream) {
  (void)in_sizes; (void)n_in; (void)out_size; (void)ws_size;
  const float* x = (const float*)d_in[0];
  const float* w_qkv = (const float*)d_in[1];
  const float* w_o = (const float*)d_in[2];
  const float* rms_w = (const float*)d_in[3];
  // d_in[4]/d_in[5] (cache_k/cache_v) fully overwritten, never returned.
  float* out = (float*)d_out;

  char* ws = (char*)d_ws;
  size_t off = 0;
  auto take = [&](size_t n) {
    char* p = ws + off;
    off += (n + 255) & ~(size_t)255;
    return p;
  };
  float* inv_sx = (float*)take(T_SEQ * sizeof(float));
  float* inv_sa = (float*)take(T_SEQ * sizeof(float));
  // aliased: xq (phase 1) / attn-quant (phase 2)
  _Float16* bufH = (_Float16*)take((size_t)T_SEQ * DIM * sizeof(_Float16));
  // aliased: xqkv f32 (phase 1) / attention output f32 (phase 2)
  float* buf32 = (float*)take((size_t)T_SEQ * QKV_N * sizeof(float));
  _Float16* Qh = (_Float16*)take((size_t)NH * T_SEQ * HD * sizeof(_Float16));
  _Float16* Kh = (_Float16*)take((size_t)NKV * T_SEQ * HD * sizeof(_Float16));
  _Float16* Vt = (_Float16*)take((size_t)NKV * HD * T_SEQ * sizeof(_Float16));
  _Float16* Wq16 = (_Float16*)take((size_t)QKV_N * DIM * sizeof(_Float16));
  _Float16* Wo16 = (_Float16*)take((size_t)DIM * DIM * sizeof(_Float16));

  cvt_f32_f16<<<(QKV_N * DIM) / 2048, 256, 0, stream>>>(w_qkv, Wq16);
  cvt_f32_f16<<<(DIM * DIM) / 2048, 256, 0, stream>>>(w_o, Wo16);
  quant_rows<<<T_SEQ, 256, 0, stream>>>(x, bufH, inv_sx);
  gemm_rowscale<<<dim3(QKV_N / 256, T_SEQ / 128), 256, 0, stream>>>(
      bufH, Wq16, inv_sx, buf32, QKV_N, DIM);
  rope_pack<<<(T_SEQ * (QKV_N / 2)) / 256, 256, 0, stream>>>(buf32, Qh, Kh, Vt);
  attn_fa<<<dim3(T_SEQ / QT, NH), 256, 0, stream>>>(Qh, Kh, Vt, buf32);
  rms_quant_rows<<<T_SEQ, 256, 0, stream>>>(buf32, rms_w, bufH, inv_sa);
  gemm_rowscale<<<dim3(DIM / 256, T_SEQ / 128), 256, 0, stream>>>(
      bufH, Wo16, inv_sa, out, DIM, DIM);
}
